// MHA_76261439308082
// MI455X (gfx1250) — matrix operand order repaired and checked by reading, not compiled, not run on hardware
//
#include <hip/hip_runtime.h>
#include <hip/hip_bf16.h>

// MHA: B=4, S=2048, D=512, H=8, HD=64.  fp32 in/out, bf16 WMMA internally.
// Workspace (~33.5 MB):
//   xb  [8192][512]        bf16   x cast
//   wtq/wtk/wtv [512][512] bf16   W transposed (Wt[k][n] = W[n][k])
//   qb  [B][H][S][64]      bf16   Q * 1/sqrt(512)
//   ktb [B][H][64][S]      bf16   K transposed (rows = head-dim)
//   vb  [B][H][S][64]      bf16   V

typedef unsigned int u32;
typedef unsigned short ushort_t;
typedef __attribute__((ext_vector_type(4)))  u32    v4u;
typedef __attribute__((ext_vector_type(16))) __bf16 v16bf;
typedef __attribute__((ext_vector_type(8)))  float  v8f;

union Frag { v4u q[2]; v16bf v; };   // 32B: one WMMA bf16 A/B operand per lane

__device__ __forceinline__ ushort_t f2bf(float f) {
    u32 u = __float_as_uint(f);
    u += 0x7FFFu + ((u >> 16) & 1u);     // round-to-nearest-even
    return (ushort_t)(u >> 16);
}

__device__ __forceinline__ v8f bwmma(v16bf a, v16bf b, v8f c) {
    return __builtin_amdgcn_wmma_f32_16x16x32_bf16(false, a, false, b,
                                                   (short)0, c, false, false);
}

// Flat shared-pointer low 32 bits == LDS byte offset.
__device__ __forceinline__ u32 ldsoff(const void* p) {
    return (u32)(unsigned long long)p;
}

// CDNA5 async DMA: global -> LDS, 16B per lane, tracked by ASYNCcnt.
__device__ __forceinline__ void async_ld16(u32 lds_addr, const void* gaddr) {
    asm volatile("global_load_async_to_lds_b128 %0, %1, off"
                 :: "v"(lds_addr), "v"(gaddr) : "memory");
}

// Wait until ASYNCcnt <= n.  Async loads complete in order, so with P ops/tile,
// "wait P" guarantees the previous tile is resident while the newest prefetch
// (P ops) stays in flight.
#define WAIT_ASYNC(n) asm volatile("s_wait_asynccnt " #n ::: "memory")

// ---------------- kernel 1: x fp32 -> bf16 ----------------
__global__ void mha_cvt_x(const float* __restrict__ in, ushort_t* __restrict__ out, int n) {
    int i = blockIdx.x * blockDim.x + threadIdx.x;
    if (i < n) out[i] = f2bf(in[i]);
}

// ---------------- kernel 2: W fp32 -> Wt bf16 (512x512 transpose) ----------------
__global__ void mha_cvt_wt(const float* __restrict__ W, ushort_t* __restrict__ Wt) {
    __shared__ float tile[32][33];
    int k0 = blockIdx.x * 32, n0 = blockIdx.y * 32;
    int tx = threadIdx.x & 31, ty = threadIdx.x >> 5;      // ty in 0..7
    #pragma unroll
    for (int i = 0; i < 32; i += 8)
        tile[ty + i][tx] = W[(n0 + ty + i) * 512 + k0 + tx];
    __syncthreads();
    #pragma unroll
    for (int i = 0; i < 32; i += 8)
        Wt[(k0 + ty + i) * 512 + n0 + tx] = f2bf(tile[tx][ty + i]);
}

// ---------------- kernel 3: fused QKV projection GEMM ----------------
// grid (M/128=64, N/64=8, 3), block 256 (8 waves); wave computes 16 rows x 64 cols.
// B tile (Wt 32x64) via async-DMA, TRIPLE buffered, threshold waits (1 op/tile/thread).
__global__ void __launch_bounds__(256, 1) mha_proj(
        const ushort_t* __restrict__ xb,
        const ushort_t* __restrict__ wtq,
        const ushort_t* __restrict__ wtk,
        const ushort_t* __restrict__ wtv,
        ushort_t* __restrict__ qb,
        ushort_t* __restrict__ ktb,
        ushort_t* __restrict__ vb) {
    const int z = blockIdx.z;
    const ushort_t* wt = (z == 0) ? wtq : (z == 1) ? wtk : wtv;

    const int lane  = threadIdx.x & 31;
    const int wave  = threadIdx.x >> 5;
    const int mbase = blockIdx.x * 128 + wave * 16;
    const int nbase = blockIdx.y * 64;
    const int r     = lane & 15;
    const int khalf = lane >> 4;

    __shared__ __align__(16) ushort_t Bt[3][32 * 72];   // padded rows, 3 buffers

    // staging map: 256 threads x one 16B chunk = 32x64 bf16 tile
    const int srow = threadIdx.x >> 3;         // 0..31
    const int scol = (threadIdx.x & 7) * 8;    // element col 0..56
    const u32  sdst = srow * 72 + scol;        // LDS element offset within buffer
    const ushort_t* sgsrc = wt + srow * 512 + nbase + scol;

    v8f acc[4] = {};
    const ushort_t* xrow = xb + (mbase + r) * 512;

    // prefetch tile 0
    async_ld16(ldsoff(&Bt[0][sdst]), sgsrc);

    int cur = 0;
    for (int ks = 0; ks < 16; ++ks) {
        const int k   = ks * 32;
        const int nxt = (cur == 2) ? 0 : cur + 1;
        if (ks + 1 < 16) {
            // prefetch tile ks+1 (target buffer last read at ks-2: fenced)
            async_ld16(ldsoff(&Bt[nxt][sdst]), sgsrc + (k + 32) * 512);
            WAIT_ASYNC(1);      // tile ks resident; tile ks+1 stays in flight
        } else {
            WAIT_ASYNC(0);
        }
        __syncthreads();

        Frag a;
        a.q[0] = *(const v4u*)&xrow[k + khalf * 8];
        a.q[1] = *(const v4u*)&xrow[k + 16 + khalf * 8];

        Frag bfr[4];
        #pragma unroll
        for (int g = 0; g < 4; ++g) {
            for (int s = 0; s < 16; ++s) { int kk = (s < 8) ? khalf * 8 + s : 16 + khalf * 8 + s - 8; ((ushort_t*)&bfr[g])[s] = Bt[cur][kk * 72 + g * 16 + r]; }
        }
        #pragma unroll
        for (int g = 0; g < 4; ++g)
            acc[g] = bwmma(a.v, bfr[g].v, acc[g]);

        cur = nxt;
    }

    const float scale = (z == 0) ? 0.04419417382415922f : 1.0f;  // 1/sqrt(512)
    #pragma unroll
    for (int g = 0; g < 4; ++g)
        #pragma unroll
        for (int e = 0; e < 8; ++e) {
            int m = mbase + e + khalf * 8;           // token index (flat B*S)
            int n = nbase + g * 16 + r;              // feature
            int bi = m >> 11, s = m & 2047;
            int h = n >> 6, d = n & 63;
            ushort_t val = f2bf(acc[g][e] * scale);
            if (z == 0)      qb [(((bi * 8 + h) * 2048) + s) * 64 + d] = val;
            else if (z == 1) ktb[(((bi * 8 + h) * 64) + d) * 2048 + s] = val;
            else             vb [(((bi * 8 + h) * 2048) + s) * 64 + d] = val;
        }
}

// ---------------- kernel 4: flash attention ----------------
// grid (S/64=32, H=8, B=4), block 128 (4 waves); wave owns 16 query rows.
// K/V tiles (64x64 bf16 each) via async-DMA, TRIPLE buffered, threshold waits
// (8 ops/tile/thread).  LDS: 3*(9216+9216)+10240 = 64 KB.
__global__ void __launch_bounds__(128, 1) mha_attn(
        const ushort_t* __restrict__ qb,
        const ushort_t* __restrict__ ktb,
        const ushort_t* __restrict__ vb,
        float* __restrict__ out) {
    const int h = blockIdx.y, b = blockIdx.z;
    const int lane  = threadIdx.x & 31;
    const int wave  = threadIdx.x >> 5;
    const int qbase = blockIdx.x * 64 + wave * 16;
    const int r     = lane & 15;
    const int khalf = lane >> 4;

    const ushort_t* Qp = qb  + ((b * 8 + h) * 2048) * 64;   // [S][64]
    const ushort_t* Kp = ktb + ((b * 8 + h) * 64) * 2048;   // [64][S]
    const ushort_t* Vp = vb  + ((b * 8 + h) * 2048) * 64;   // [S][64]

    __shared__ __align__(16) ushort_t Ks[3][64 * 72];       // [d][key], pad 72
    __shared__ __align__(16) ushort_t Vs[3][64 * 72];       // [key][d], pad 72
    __shared__ __align__(16) ushort_t Ps[4 * 16 * 80];      // per-wave P, pad 80
    ushort_t* Pw = &Ps[wave * 16 * 80];

    // staging map: 128 threads, 4 chunks each per K and V tile
    const int srow0 = threadIdx.x >> 3;        // 0..15 (+16*i)
    const int scol  = (threadIdx.x & 7) * 8;

    // Q fragments (A-layout), loaded once
    Frag Q0, Q1;
    {
        const ushort_t* qrow = Qp + (qbase + r) * 64;
        Q0.q[0] = *(const v4u*)&qrow[khalf * 8];
        Q0.q[1] = *(const v4u*)&qrow[16 + khalf * 8];
        Q1.q[0] = *(const v4u*)&qrow[32 + khalf * 8];
        Q1.q[1] = *(const v4u*)&qrow[48 + khalf * 8];
    }

    float mrow[8], lrow[8];
    v8f o[4] = {};
    #pragma unroll
    for (int e = 0; e < 8; ++e) { mrow[e] = -3.0e38f; lrow[e] = 0.0f; }

    // prefetch key-block 0
    #pragma unroll
    for (int i = 0; i < 4; ++i) {
        int row = srow0 + i * 16;
        async_ld16(ldsoff(&Ks[0][row * 72 + scol]), &Kp[row * 2048 + scol]);
        async_ld16(ldsoff(&Vs[0][row * 72 + scol]), &Vp[row * 64 + scol]);
    }

    int cur = 0;
    for (int kb = 0; kb < 2048; kb += 64) {
        const int nxt = (cur == 2) ? 0 : cur + 1;
        if (kb + 64 < 2048) {
            // prefetch next block (target buffers last read 2 iterations ago)
            #pragma unroll
            for (int i = 0; i < 4; ++i) {
                int row = srow0 + i * 16;
                async_ld16(ldsoff(&Ks[nxt][row * 72 + scol]),
                           &Kp[row * 2048 + kb + 64 + scol]);
                async_ld16(ldsoff(&Vs[nxt][row * 72 + scol]),
                           &Vp[(kb + 64 + row) * 64 + scol]);
            }
            WAIT_ASYNC(8);      // block kb resident; prefetch stays in flight
        } else {
            WAIT_ASYNC(0);
        }
        __syncthreads();

        // scores: S = Q @ K^T   (16 q x 64 keys, f32 acc)
        v8f sc[4] = {};
        {
            Frag bk[4];
            #pragma unroll
            for (int g = 0; g < 4; ++g) {
                for (int s = 0; s < 16; ++s) { int kk = (s < 8) ? khalf * 8 + s : 16 + khalf * 8 + s - 8; ((ushort_t*)&bk[g])[s] = Ks[cur][kk * 72 + g * 16 + r]; }
            }
            #pragma unroll
            for (int g = 0; g < 4; ++g) sc[g] = bwmma(Q0.v, bk[g].v, sc[g]);
            #pragma unroll
            for (int g = 0; g < 4; ++g) {
                for (int s = 0; s < 16; ++s) { int kk = (s < 8) ? khalf * 8 + s : 16 + khalf * 8 + s - 8; ((ushort_t*)&bk[g])[s] = Ks[cur][(32 + kk) * 72 + g * 16 + r]; }
            }
            #pragma unroll
            for (int g = 0; g < 4; ++g) sc[g] = bwmma(Q1.v, bk[g].v, sc[g]);
        }

        // online softmax (row = e + 8*khalf, spread over 16 lanes)
        #pragma unroll
        for (int e = 0; e < 8; ++e) {
            float mx = fmaxf(fmaxf(sc[0][e], sc[1][e]), fmaxf(sc[2][e], sc[3][e]));
            #pragma unroll
            for (int s = 8; s >= 1; s >>= 1) mx = fmaxf(mx, __shfl_xor(mx, s, 32));
            float mnew = fmaxf(mrow[e], mx);
            float corr = exp2f((mrow[e] - mnew) * 1.44269504f);
            mrow[e] = mnew;
            float rs = 0.0f;
            float p[4];
            #pragma unroll
            for (int g = 0; g < 4; ++g) {
                p[g] = exp2f((sc[g][e] - mnew) * 1.44269504f);
                rs += p[g];
            }
            #pragma unroll
            for (int s = 8; s >= 1; s >>= 1) rs += __shfl_xor(rs, s, 32);
            lrow[e] = lrow[e] * corr + rs;
            #pragma unroll
            for (int g = 0; g < 4; ++g) {
                o[g][e] *= corr;
                // C-layout -> LDS (row-major 16x64, stride 80)
                Pw[(e + khalf * 8) * 80 + g * 16 + r] = f2bf(p[g]);
            }
        }

        // reload P in A-layout (per-wave LDS region; DS ops in-order per wave)
        Frag P0, P1;
        P0.q[0] = *(const v4u*)&Pw[r * 80 + khalf * 8];
        P0.q[1] = *(const v4u*)&Pw[r * 80 + 16 + khalf * 8];
        P1.q[0] = *(const v4u*)&Pw[r * 80 + 32 + khalf * 8];
        P1.q[1] = *(const v4u*)&Pw[r * 80 + 48 + khalf * 8];

        // O += P @ V
        {
            Frag bv[4];
            #pragma unroll
            for (int g = 0; g < 4; ++g) {
                for (int s = 0; s < 16; ++s) { int kk = (s < 8) ? khalf * 8 + s : 16 + khalf * 8 + s - 8; ((ushort_t*)&bv[g])[s] = Vs[cur][kk * 72 + g * 16 + r]; }
            }
            #pragma unroll
            for (int g = 0; g < 4; ++g) o[g] = bwmma(P0.v, bv[g].v, o[g]);
            #pragma unroll
            for (int g = 0; g < 4; ++g) {
                for (int s = 0; s < 16; ++s) { int kk = (s < 8) ? khalf * 8 + s : 16 + khalf * 8 + s - 8; ((ushort_t*)&bv[g])[s] = Vs[cur][(32 + kk) * 72 + g * 16 + r]; }
            }
            #pragma unroll
            for (int g = 0; g < 4; ++g) o[g] = bwmma(P1.v, bv[g].v, o[g]);
        }

        cur = nxt;
    }

    // epilogue: normalize and write fp32 output [B][S][H*64]
    #pragma unroll
    for (int e = 0; e < 8; ++e) {
        float inv = 1.0f / lrow[e];
        int q = qbase + e + khalf * 8;
        float* orow = out + (b * 2048 + q) * 512 + h * 64;
        #pragma unroll
        for (int g = 0; g < 4; ++g)
            orow[g * 16 + r] = o[g][e] * inv;
    }
}

extern "C" void kernel_launch(void* const* d_in, const int* in_sizes, int n_in,
                              void* d_out, int out_size, void* d_ws, size_t ws_size,
                              hipStream_t stream) {
    const float* x  = (const float*)d_in[0];
    const float* Wq = (const float*)d_in[1];
    const float* Wk = (const float*)d_in[2];
    const float* Wv = (const float*)d_in[3];
    float* out = (float*)d_out;

    char* ws = (char*)d_ws;
    ushort_t* xb  = (ushort_t*)ws;                 // 8192*512
    ushort_t* wtq = (ushort_t*)(ws + 8388608);     // 512*512
    ushort_t* wtk = wtq + 262144;
    ushort_t* wtv = wtk + 262144;
    ushort_t* qb  = wtv + 262144;                  // 4*8*2048*64 each
    ushort_t* ktb = qb + 4194304;
    ushort_t* vb  = ktb + 4194304;

    mha_cvt_x<<<16384, 256, 0, stream>>>(x, xb, 4194304);
    dim3 tg(16, 16);
    mha_cvt_wt<<<tg, 256, 0, stream>>>(Wq, wtq);
    mha_cvt_wt<<<tg, 256, 0, stream>>>(Wk, wtk);
    mha_cvt_wt<<<tg, 256, 0, stream>>>(Wv, wtv);
    mha_proj<<<dim3(64, 8, 3), 256, 0, stream>>>(xb, wtq, wtk, wtv, qb, ktb, vb);
    mha_attn<<<dim3(32, 8, 4), 128, 0, stream>>>(qb, ktb, vb, out);
}
